// EdgeConv_56195352101070
// MI455X (gfx1250) — matrix operand order repaired and checked by reading, not compiled, not run on hardware
//
#include <hip/hip_runtime.h>

typedef __attribute__((ext_vector_type(16))) _Float16 v16h;
typedef __attribute__((ext_vector_type(8)))  _Float16 v8h;
typedef __attribute__((ext_vector_type(8)))  float    v8f;
typedef __attribute__((ext_vector_type(4)))  unsigned int v4u;
typedef __attribute__((ext_vector_type(8)))  unsigned int v8u;

#define B_CFG 16
#define N_CFG 128
#define F_CFG 100
#define H_CFG 128
#define PITCH 136   // 128 + 8 halves row padding: 68 dwords -> bank-quad spread

// ---------------------------------------------------------------------------
// Prep kernel 1: u[b,n,h] = sum_f x[b,n,f]*(W1a-W1b)[f,h] + b1[h]
//                v[b,n,h] = sum_f x[b,n,f]*W1b[f,h]
// ---------------------------------------------------------------------------
__global__ __launch_bounds__(128)
void edgeconv_uv_kernel(const float* __restrict__ x,
                        const float* __restrict__ W1,
                        const float* __restrict__ b1,
                        float* __restrict__ u,
                        float* __restrict__ v) {
  __shared__ float sx[F_CFG];
  const int bn = blockIdx.x;      // b*128 + n
  const int h  = threadIdx.x;     // 0..127
  if (h < F_CFG) sx[h] = x[(size_t)bn * F_CFG + h];
  __syncthreads();
  float au = 0.f, av = 0.f;
  #pragma unroll 4
  for (int f = 0; f < F_CFG; ++f) {
    const float xv = sx[f];
    const float wa = W1[f * H_CFG + h];            // W1a row
    const float wb = W1[(F_CFG + f) * H_CFG + h];  // W1b row
    au = fmaf(xv, wa - wb, au);
    av = fmaf(xv, wb, av);
  }
  u[(size_t)bn * H_CFG + h] = au + b1[h];
  v[(size_t)bn * H_CFG + h] = av;
}

// ---------------------------------------------------------------------------
// Prep kernel 2: W2T[k][h] = (f16) W2[h][k]  (row-major, unpadded in global;
// the TDM adds the LDS padding during the DMA)
// ---------------------------------------------------------------------------
__global__ __launch_bounds__(256)
void edgeconv_w2t_kernel(const float* __restrict__ W2,
                         _Float16* __restrict__ w2t) {
  const int idx = blockIdx.x * 256 + threadIdx.x;  // 0..16383
  const int k = idx >> 7;
  const int h = idx & 127;
  w2t[idx] = (_Float16)W2[h * H_CFG + k];
}

// ---------------------------------------------------------------------------
// Main fused kernel: one workgroup per (b,i). 8 waves x 16 rows.
//   H1[j][h] = relu(u[b,i,h] + v[b,j,h])             (f16, LDS)
//   H2 = relu(H1 @ W2 + b2)                          (v_wmma_f32_16x16x32_f16)
//   out[b,i,k] = max_j H2[j][k] * adj[b,i,j]
// W2^T is staged into LDS by the Tensor Data Mover (tensor_load_to_lds),
// using TDM pad_enable to emit the 68-dword padded row pitch.
// ---------------------------------------------------------------------------
__global__ __launch_bounds__(256)
void edgeconv_main_kernel(const float* __restrict__ u,
                          const float* __restrict__ v,
                          const _Float16* __restrict__ w2t,
                          const float* __restrict__ b2,
                          const float* __restrict__ adj,
                          float* __restrict__ out) {
  __shared__ _Float16 sH1[N_CFG * PITCH];   // ~34 KB, later reused as f32 red
  __shared__ _Float16 sW2[H_CFG * PITCH];   // ~34 KB (TDM destination)

  const int bi   = blockIdx.x;         // b*128 + i
  const int b    = bi >> 7;
  const int t    = threadIdx.x;        // 0..255
  const int lane = t & 31;
  const int wave = t >> 5;
  const int l15  = lane & 15;
  const int lsel = lane >> 4;          // 0: lanes 0-15, 1: lanes 16-31

  // --- TDM: DMA W2^T (128x128 f16) global -> LDS with row padding ---------
  if (wave == 0) {
    typedef __attribute__((address_space(3))) _Float16 lds_h;
    const unsigned int lds_off = (unsigned int)(unsigned long long)(lds_h*)sW2;
    const unsigned long long ga = (unsigned long long)w2t;
    // D# group 0: count=1 | lds_addr | global_addr[56:0] | type=2
    v4u g0;
    g0[0] = 1u;                                    // count=1, is_restore=0
    g0[1] = lds_off;                               // lds_addr (bytes)
    g0[2] = (unsigned int)ga;                      // global_addr[31:0]
    g0[3] = (unsigned int)((ga >> 32) & 0x01FFFFFFu) | 0x80000000u; // [56:32], type=2
    // D# group 1
    v8u g1;
    g1[0] = (1u << 16)        // data_size = 1 (2-byte elements)
          | (1u << 20)        // pad_enable
          | (5u << 22)        // pad_interval: 64 dwords (2^(5+1))
          | (3u << 25);       // pad_amount: 4 dwords (3+1) -> 68-dword pitch
    g1[1] = (128u << 16);     // tensor_dim0[15:0] = 128 (bits 79:48)
    g1[2] = (128u << 16);     // tensor_dim0[31:16]=0; tensor_dim1[15:0]=128
    g1[3] = (128u << 16);     // tensor_dim1[31:16]=0; tile_dim0 = 128
    g1[4] = 128u;             // tile_dim1 = 128; tile_dim2 = 0
    g1[5] = 128u;             // tensor_dim0_stride[31:0] = 128 elements
    g1[6] = 0u;               // stride0 hi, tensor_dim1_stride lo (unused)
    g1[7] = 0u;
    v4u g2 = {0u, 0u, 0u, 0u};
    v4u g3 = {0u, 0u, 0u, 0u};
    asm volatile("tensor_load_to_lds %0, %1, %2, %3"
                 :: "s"(g0), "s"(g1), "s"(g2), "s"(g3) : "memory");
  }

  // --- build H1 = relu(u_i + v_j) in f16: thread -> (row j, 64-col half) --
  {
    const int j  = t >> 1;
    const int hb = (t & 1) * 64;
    const float*  up = u + (size_t)bi * H_CFG + hb;
    const float*  vp = v + ((size_t)(b * N_CFG + j)) * H_CFG + hb;
    _Float16*     hp = sH1 + j * PITCH + hb;
    #pragma unroll
    for (int c = 0; c < 64; c += 8) {
      float4 u0 = *(const float4*)(up + c);
      float4 u1 = *(const float4*)(up + c + 4);
      float4 v0 = *(const float4*)(vp + c);
      float4 v1 = *(const float4*)(vp + c + 4);
      v8h pk;
      pk[0] = (_Float16)fmaxf(u0.x + v0.x, 0.f);
      pk[1] = (_Float16)fmaxf(u0.y + v0.y, 0.f);
      pk[2] = (_Float16)fmaxf(u0.z + v0.z, 0.f);
      pk[3] = (_Float16)fmaxf(u0.w + v0.w, 0.f);
      pk[4] = (_Float16)fmaxf(u1.x + v1.x, 0.f);
      pk[5] = (_Float16)fmaxf(u1.y + v1.y, 0.f);
      pk[6] = (_Float16)fmaxf(u1.z + v1.z, 0.f);
      pk[7] = (_Float16)fmaxf(u1.w + v1.w, 0.f);
      *(v8h*)(hp + c) = pk;
    }
  }
  if (wave == 0) __builtin_amdgcn_s_wait_tensorcnt(0);
  __syncthreads();

  // --- WMMA GEMM: this wave computes rows [16*wave, 16*wave+16) x 128 cols -
  const int rowbase = wave * 16;
  v8f acc[8] = {};   // 8 N-tiles of 16x16 f32

  #pragma unroll
  for (int k = 0; k < 4; ++k) {
    const int kbase = k * 32;
    // A fragment (16x32 f16): lane l<16 -> row rowbase+l, K {0..15};
    //                         lane l>=16 -> row rowbase+l-16, K {16..31}
    const _Float16* ap = sH1 + (rowbase + l15) * PITCH + kbase + lsel * 16;
    v8h a0 = *(const v8h*)ap;          // K = kbase + lsel*16 + 0..7
    v8h a1 = *(const v8h*)(ap + 8);    // K = kbase + lsel*16 + 8..15
    v16h afrag;
    #pragma unroll
    for (int e = 0; e < 8; ++e) { afrag[e] = a0[e]; afrag[e + 8] = a1[e]; }

    // Batch-load all 8 B fragments for this K-step, then issue 8 WMMAs
    // back-to-back (collapses ds waits to one per K-step).
    v16h bfr[8];
    #pragma unroll
    for (int n = 0; n < 8; ++n)
      bfr[n] = *(const v16h*)(sW2 + (n * 16 + l15) * PITCH + kbase + lsel * 16);
    #pragma unroll
    for (int n = 0; n < 8; ++n)
      acc[n] = __builtin_amdgcn_wmma_f32_16x16x32_f16(
          false, afrag, false, bfr[n], (short)0, acc[n], false, false);
  }

  // --- epilogue: relu(+b2), * adj[b,i,j], max over j ----------------------
  // C layout: acc[n][r] is element (row = rowbase + lsel*8 + r, col = n*16+l15)
  const float* adjrow = adj + (size_t)bi * N_CFG;
  float adjv[8];
  #pragma unroll
  for (int r = 0; r < 8; ++r) adjv[r] = adjrow[rowbase + lsel * 8 + r];

  float pm[8];
  #pragma unroll
  for (int n = 0; n < 8; ++n) {
    const float bias = b2[n * 16 + l15];
    float m = 0.f;   // all candidates are >= 0 (relu * nonneg adj)
    #pragma unroll
    for (int r = 0; r < 8; ++r) {
      const float val = fmaxf(acc[n][r] + bias, 0.f) * adjv[r];
      m = fmaxf(m, val);
    }
    m = fmaxf(m, __shfl_xor(m, 16, 32));  // fold partner half-lane rows
    pm[n] = m;
  }

  __syncthreads();                       // done reading sH1; reuse as f32 red
  float* red = (float*)sH1;              // [8 waves][128 cols]
  if (lsel == 0) {
    #pragma unroll
    for (int n = 0; n < 8; ++n) red[wave * 128 + n * 16 + l15] = pm[n];
  }
  __syncthreads();

  if (t < 128) {
    float m = red[t];
    #pragma unroll
    for (int w = 1; w < 8; ++w) m = fmaxf(m, red[w * 128 + t]);
    out[(size_t)bi * H_CFG + t] = m;
  }
}

// ---------------------------------------------------------------------------
// Launch
// ---------------------------------------------------------------------------
extern "C" void kernel_launch(void* const* d_in, const int* in_sizes, int n_in,
                              void* d_out, int out_size, void* d_ws, size_t ws_size,
                              hipStream_t stream) {
  const float* x   = (const float*)d_in[0];  // (16,128,100)
  const float* adj = (const float*)d_in[1];  // (16,128,128)
  const float* W1  = (const float*)d_in[2];  // (200,128)
  const float* b1  = (const float*)d_in[3];  // (128)
  const float* W2  = (const float*)d_in[4];  // (128,128)
  const float* b2  = (const float*)d_in[5];  // (128)
  float* out = (float*)d_out;                // (16,128,128)

  char* ws = (char*)d_ws;
  float*    u   = (float*)(ws);                       // 1 MB
  float*    v   = (float*)(ws + (1u << 20));          // 1 MB
  _Float16* w2t = (_Float16*)(ws + (2u << 20));       // 32 KB

  edgeconv_uv_kernel<<<dim3(B_CFG * N_CFG), dim3(128), 0, stream>>>(x, W1, b1, u, v);
  edgeconv_w2t_kernel<<<dim3(64), dim3(256), 0, stream>>>(W2, w2t);
  edgeconv_main_kernel<<<dim3(B_CFG * N_CFG), dim3(256), 0, stream>>>(u, v, w2t, b2, adj, out);
}
